// OGRENet_73959336837504
// MI455X (gfx1250) — compile-verified, matrix operand order checked by reading, not run on hardware
//
#include <hip/hip_runtime.h>

// ---------------------------------------------------------------------------
// OGRENet on MI455X (gfx1250): bf16 WMMA GEMM pipeline with TDM tile staging.
// ---------------------------------------------------------------------------

#define N_NODES  16384
#define N_EDGES  131072
#define N_GRAPHS 16
#define U_SZ     4096
#define U_RED    256
#define NODE_IN  9

typedef __attribute__((ext_vector_type(16))) __bf16        v16bf;
typedef __attribute__((ext_vector_type(8)))  float         v8f;
typedef __attribute__((ext_vector_type(4)))  unsigned int  v4u;
typedef __attribute__((ext_vector_type(8)))  int           v8i;
typedef __attribute__((ext_vector_type(4)))  int           v4i;

__device__ __forceinline__ unsigned short f2bf(float f) {
  unsigned int u = __float_as_uint(f);
  unsigned int r = u + 0x7fffu + ((u >> 16) & 1u);   // round-to-nearest-even
  return (unsigned short)(r >> 16);
}
__device__ __forceinline__ float bf2f(unsigned short h) {
  return __uint_as_float(((unsigned int)h) << 16);
}

// ---------------------------------------------------------------------------
// u_r = u @ w_dr + b_dr        [16,4096]x[4096,256] -> [16,256]  (tiny, fp32)
// ---------------------------------------------------------------------------
__global__ void k_u_reduce(const float* __restrict__ u,
                           const float* __restrict__ w,
                           const float* __restrict__ b,
                           float* __restrict__ ur) {
  int g = blockIdx.x;
  int j = threadIdx.x;
  const float* ug = u + g * U_SZ;
  float s = b[j];
  for (int k = 0; k < U_SZ; ++k) s = fmaf(ug[k], w[k * U_RED + j], s);
  ur[g * U_RED + j] = s;
}

// ---------------------------------------------------------------------------
// Weight convert: W fp32 [K][N] row-major -> WT bf16 [N][Kpad], zero padded.
// ---------------------------------------------------------------------------
__global__ void k_wT(const float* __restrict__ W, unsigned short* __restrict__ WT,
                     int K, int N, int Kpad) {
  long long i = (long long)blockIdx.x * blockDim.x + threadIdx.x;
  long long total = (long long)N * Kpad;
  if (i >= total) return;
  int n = (int)(i / Kpad);
  int k = (int)(i % Kpad);
  WT[i] = (k < K) ? f2bf(W[(long long)k * N + n]) : (unsigned short)0;
}

// ---------------------------------------------------------------------------
// e_in bf16 [E][288]: [x[row](9) | x[col](9) | edge_attr(1) | u_r[batch[row]](256) | 0..]
// ---------------------------------------------------------------------------
__global__ void k_build_e_in(const float* __restrict__ x,
                             const float* __restrict__ ea,
                             const float* __restrict__ ur,
                             const int* __restrict__ ei,
                             const int* __restrict__ batch,
                             unsigned short* __restrict__ out) {
  int e = blockIdx.x;
  int j = threadIdx.x;
  int r = ei[e];
  int c = ei[N_EDGES + e];
  float v;
  if (j < 9)        v = x[r * NODE_IN + j];
  else if (j < 18)  v = x[c * NODE_IN + (j - 9)];
  else if (j == 18) v = ea[e];
  else if (j < 275) v = ur[batch[r] * U_RED + (j - 19)];
  else              v = 0.0f;
  out[(long long)e * 288 + j] = f2bf(v);
}

// ---------------------------------------------------------------------------
// n_in bf16 [E][544]: cols 0..8 = x[col], cols 521..543 = 0.
// (cols 9..520 are written by the edge-final GEMM with cofs=9.)
// ---------------------------------------------------------------------------
__global__ void k_fill_n_in(const float* __restrict__ x,
                            const int* __restrict__ ei,
                            unsigned short* __restrict__ nin) {
  int e = blockIdx.x;
  int j = threadIdx.x;          // 0..31
  long long base = (long long)e * 544;
  if (j < 9) {
    int c = ei[N_EDGES + e];
    nin[base + j] = f2bf(x[c * NODE_IN + j]);
  } else {
    nin[base + 521 + (j - 9)] = 0;  // j-9 in 0..22 -> cols 521..543
  }
}

__global__ void k_zero(float* __restrict__ p, long long n) {
  long long i = (long long)blockIdx.x * blockDim.x + threadIdx.x;
  if (i < n) p[i] = 0.0f;
}

// ---------------------------------------------------------------------------
// Scatter-sum h2[E][512] (bf16) by row index into agg[N][512] fp32 + cnt[N].
// ---------------------------------------------------------------------------
__global__ void k_scatter(const unsigned short* __restrict__ h2,
                          const int* __restrict__ ei,
                          float* __restrict__ agg,
                          float* __restrict__ cnt) {
  int e = blockIdx.x;
  int t = threadIdx.x;
  int r = ei[e];
  const unsigned short* h = h2 + (long long)e * 512;
  float* a = agg + (long long)r * 512;
  atomicAdd(&a[t],       bf2f(h[t]));
  atomicAdd(&a[t + 256], bf2f(h[t + 256]));
  if (t == 0) atomicAdd(&cnt[r], 1.0f);
}

// ---------------------------------------------------------------------------
// n2_in bf16 [N][800]: [x(9) | agg/max(cnt,1) (512) | u_r[batch] (256) | 0..]
// ---------------------------------------------------------------------------
__global__ void k_build_n2_in(const float* __restrict__ x,
                              const float* __restrict__ agg,
                              const float* __restrict__ cnt,
                              const float* __restrict__ ur,
                              const int* __restrict__ batch,
                              unsigned short* __restrict__ out) {
  int n = blockIdx.x;
  float inv = 1.0f / fmaxf(cnt[n], 1.0f);
  const float* an = agg + (long long)n * 512;
  int ub = batch[n] * U_RED;
  for (int j = threadIdx.x; j < 800; j += 256) {
    float v;
    if (j < 9)        v = x[n * NODE_IN + j];
    else if (j < 521) v = an[j - 9] * inv;
    else if (j < 777) v = ur[ub + (j - 521)];
    else              v = 0.0f;
    out[(long long)n * 800 + j] = f2bf(v);
  }
}

// ---------------------------------------------------------------------------
// Final: out[n] = dot(h3[n][0..511], n2_w1) + n2_b1   (fp32 out)
// ---------------------------------------------------------------------------
__global__ void k_final(const unsigned short* __restrict__ h3,
                        const float* __restrict__ w,
                        const float* __restrict__ b,
                        float* __restrict__ out) {
  int n = blockIdx.x * blockDim.x + threadIdx.x;
  if (n >= N_NODES) return;
  const unsigned short* h = h3 + (long long)n * 512;
  float s = b[0];
  for (int c = 0; c < 512; ++c) s = fmaf(bf2f(h[c]), w[c], s);
  out[n] = s;
}

// ---------------------------------------------------------------------------
// TDM: issue tensor_load_to_lds for one 128-row x 64-byte tile.
//   - data_size = 4 bytes; tile_dim0 = 16 DW (64B = 32 bf16); tile_dim1 = 128
//   - pad_enable: interval 16 DW (one row), amount 4 DW (16B)
//       => LDS row stride = 80B = 40 ushorts (LDSTR), bank-skewed + aligned
//   - groups 2/3 zero (2-D tile), tensor dims large (tiles always in-bounds)
// Descriptor bit packing per CDNA5 ISA 8.3/8.4.
// ---------------------------------------------------------------------------
__device__ __forceinline__ void tdm_load_tile(unsigned lds_off,
                                              const unsigned short* gptr,
                                              int row_stride_dw) {
  unsigned long long ga = (unsigned long long)(size_t)gptr;
  v4u g0;
  g0.x = 1u;                                    // count=1, user mode, no gather
  g0.y = lds_off;                               // LDS byte address
  g0.z = (unsigned)ga;                          // global_addr[31:0]
  g0.w = (unsigned)((ga >> 32) & 0x1ffffffu)    // global_addr[56:32]
         | (2u << 30);                          // type = 2 ("image")
  v8i g1;
  g1[0] = (int)((2u << 16)        // data_size = 4B
                | (1u << 20)      // pad_enable
                | (3u << 22)      // pad_interval: 16 DW
                | (3u << 25));    // pad_amount: 4 DW
  g1[1] = (int)0xffff0000u;       // tensor_dim0 = 0x7fffffff (low 16 in d1 hi)
  g1[2] = (int)0xffff7fffu;       // tensor_dim0 hi16 | tensor_dim1 low16
  g1[3] = (int)((16u << 16) | 0x7fffu);  // tile_dim0=16 DW | tensor_dim1 hi16
  g1[4] = 128;                    // tile_dim1 = 128 rows, tile_dim2 = 0
  g1[5] = row_stride_dw;          // tensor_dim0_stride[31:0] (DW units)
  g1[6] = 0;                      // stride hi | tensor_dim1_stride lo
  g1[7] = 0;
  v4i z4 = {0, 0, 0, 0};
#if defined(__clang_major__) && (__clang_major__ >= 23)
  v8i z8 = {0, 0, 0, 0, 0, 0, 0, 0};
  __builtin_amdgcn_tensor_load_to_lds(g0, g1, z4, z4, z8, 0);
#else
  __builtin_amdgcn_tensor_load_to_lds(g0, g1, z4, z4, 0);
#endif
}

// ---------------------------------------------------------------------------
// bf16 WMMA GEMM:  C[M][N] = act(A[M][K] * B[K][N] + bias)
//   A   : bf16 row-major [M][lda]           (lda = Kpad, mult of 8)
//   BT  : bf16 row-major [N][ldb] = B^T     (ldb = Kpad, mult of 8)
// Workgroup tile 128x128, 8 waves (wave32), each wave 32x64 = 2x4 WMMA tiles.
// Double-buffered LDS staged by the Tensor Data Mover: wave 0 issues the next
// A/BT tile DMA, computes, s_wait_tensorcnt, then one barrier per K-stage.
// ---------------------------------------------------------------------------
#define LDSTR 40   // ushort stride for LDS rows (80B via TDM pad)

__global__ __launch_bounds__(256) void k_gemm_bf16(
    const unsigned short* __restrict__ A,  int lda,
    const unsigned short* __restrict__ BT, int ldb,
    const float* __restrict__ bias,
    unsigned short* __restrict__ C, long long ldc, int cofs,
    int M, int N, int K, int relu)
{
  __shared__ __align__(16) unsigned short As[2][128 * LDSTR];
  __shared__ __align__(16) unsigned short Bs[2][128 * LDSTR];

  const int tid  = threadIdx.x;
  const int lane = tid & 31;
  const int w    = tid >> 5;
  const int wm   = w & 3;        // 4 waves along M
  const int wn   = w >> 2;       // 2 waves along N
  const long long m0 = (long long)blockIdx.y * 128;
  const int n0 = blockIdx.x * 128;
  const int l15   = lane & 15;
  const int khalf = (lane >> 4) << 3;   // 0 or 8 (A/B k-half, C/D m-half)

  const unsigned ldsA0 = (unsigned)(size_t)(void*)&As[0][0];
  const unsigned ldsA1 = (unsigned)(size_t)(void*)&As[1][0];
  const unsigned ldsB0 = (unsigned)(size_t)(void*)&Bs[0][0];
  const unsigned ldsB1 = (unsigned)(size_t)(void*)&Bs[1][0];
  const unsigned short* Abase = A  + m0 * lda;
  const unsigned short* Bbase = BT + (long long)n0 * ldb;

  v8f acc[2][4];
  {
    v8f z = {};
    #pragma unroll
    for (int i = 0; i < 2; ++i)
      #pragma unroll
      for (int j = 0; j < 4; ++j) acc[i][j] = z;
  }

  const int kTiles = K >> 5;

  // ---- prologue: TDM-stage tile 0 into buffer 0 ----
  if (w == 0) {
    tdm_load_tile(ldsA0, Abase, lda >> 1);
    tdm_load_tile(ldsB0, Bbase, ldb >> 1);
    __builtin_amdgcn_s_wait_tensorcnt(0);
  }
  __syncthreads();

  for (int kt = 0; kt < kTiles; ++kt) {
    const int buf = kt & 1;

    // ---- async: TDM-stage next tile into the other buffer ----
    if (w == 0 && (kt + 1) < kTiles) {
      unsigned la = (buf ? ldsA0 : ldsA1);
      unsigned lb = (buf ? ldsB0 : ldsB1);
      tdm_load_tile(la, Abase + ((kt + 1) << 5), lda >> 1);
      tdm_load_tile(lb, Bbase + ((kt + 1) << 5), ldb >> 1);
    }

    // ---- fragments per 16-bit A-matrix lane layout ----
    // lane<16: K {0..7,16..23}; lane>=16: K {8..15,24..31}
    const unsigned short* Asb = &As[buf][0];
    const unsigned short* Bsb = &Bs[buf][0];
    v16bf af[2], bfr[4];
    #pragma unroll
    for (int mi = 0; mi < 2; ++mi) {
      int row = (wm << 5) + (mi << 4) + l15;
      const uint4* p = (const uint4*)(Asb + row * LDSTR + khalf);
      union { uint4 u[2]; v16bf v; } t;
      t.u[0] = p[0];       // khalf + 0..7
      t.u[1] = p[2];       // khalf + 16..23
      af[mi] = t.v;
    }
    #pragma unroll
    for (int ni = 0; ni < 4; ++ni) {
      int row = (wn << 6) + (ni << 4) + l15;
      const uint4* p = (const uint4*)(Bsb + row * LDSTR + khalf);
      union { uint4 u[2]; v16bf v; } t;
      t.u[0] = p[0];
      t.u[1] = p[2];
      bfr[ni] = t.v;
    }

    // ---- 8 WMMAs per wave per K-stage ----
    #pragma unroll
    for (int mi = 0; mi < 2; ++mi)
      #pragma unroll
      for (int ni = 0; ni < 4; ++ni)
        acc[mi][ni] = __builtin_amdgcn_wmma_f32_16x16x32_bf16(
            false, af[mi], false, bfr[ni], (short)0, acc[mi][ni], false, false);

    // ---- wave0: ensure next tile landed; then one barrier per stage ----
    if (w == 0 && (kt + 1) < kTiles) __builtin_amdgcn_s_wait_tensorcnt(0);
    __syncthreads();
  }

  // ---- epilogue: bias (+ReLU), fp32 -> bf16, scatter per C/D lane layout ----
  #pragma unroll
  for (int mi = 0; mi < 2; ++mi) {
    #pragma unroll
    for (int ni = 0; ni < 4; ++ni) {
      int n = n0 + (wn << 6) + (ni << 4) + l15;
      float bn = bias[n];
      #pragma unroll
      for (int r = 0; r < 8; ++r) {
        long long m = m0 + (wm << 5) + (mi << 4) + khalf + r;
        float v = acc[mi][ni][r] + bn;
        if (relu) v = fmaxf(v, 0.0f);
        C[m * ldc + cofs + n] = f2bf(v);
      }
    }
  }
}

// ---------------------------------------------------------------------------
// Host side
// ---------------------------------------------------------------------------
extern "C" void kernel_launch(void* const* d_in, const int* in_sizes, int n_in,
                              void* d_out, int out_size, void* d_ws, size_t ws_size,
                              hipStream_t stream) {
  (void)in_sizes; (void)n_in; (void)out_size; (void)ws_size;

  const float* x    = (const float*)d_in[0];
  const float* ea   = (const float*)d_in[1];
  const float* u    = (const float*)d_in[2];
  const int*   ei   = (const int*)  d_in[3];
  const int*   bat  = (const int*)  d_in[4];
  const float* w_dr = (const float*)d_in[5];
  const float* b_dr = (const float*)d_in[6];
  const float* e_w0 = (const float*)d_in[7];
  const float* e_b0 = (const float*)d_in[8];
  const float* e_w1 = (const float*)d_in[9];
  const float* e_b1 = (const float*)d_in[10];
  const float* e_w2 = (const float*)d_in[11];
  const float* e_b2 = (const float*)d_in[12];
  const float* e_w3 = (const float*)d_in[13];
  const float* e_b3 = (const float*)d_in[14];
  const float* e_wf = (const float*)d_in[15];
  const float* e_bf = (const float*)d_in[16];
  const float* n1w0 = (const float*)d_in[17];
  const float* n1b0 = (const float*)d_in[18];
  const float* n1w1 = (const float*)d_in[19];
  const float* n1b1 = (const float*)d_in[20];
  const float* n2w0 = (const float*)d_in[21];
  const float* n2b0 = (const float*)d_in[22];
  const float* n2w1 = (const float*)d_in[23];
  const float* n2b1 = (const float*)d_in[24];
  float* out = (float*)d_out;

  // ---- workspace allocator (256B aligned) ----
  size_t off = 0;
  char* base = (char*)d_ws;
  auto alloc = [&](size_t bytes) -> void* {
    off = (off + 255) & ~(size_t)255;
    void* p = base + off;
    off += bytes;
    return p;
  };

  float*          ur    = (float*)         alloc(N_GRAPHS * U_RED * 4);
  unsigned short* w0T   = (unsigned short*)alloc((size_t)1024 * 288 * 2);
  unsigned short* w1T   = (unsigned short*)alloc((size_t)1024 * 1024 * 2);
  unsigned short* w2T   = (unsigned short*)alloc((size_t)1024 * 1024 * 2);
  unsigned short* w3T   = (unsigned short*)alloc((size_t)1024 * 1024 * 2);
  unsigned short* wfT   = (unsigned short*)alloc((size_t)512 * 1024 * 2);
  unsigned short* n1w0T = (unsigned short*)alloc((size_t)512 * 544 * 2);
  unsigned short* n1w1T = (unsigned short*)alloc((size_t)512 * 512 * 2);
  unsigned short* n2w0T = (unsigned short*)alloc((size_t)512 * 800 * 2);
  unsigned short* e_in  = (unsigned short*)alloc((size_t)N_EDGES * 288 * 2);
  unsigned short* h_a   = (unsigned short*)alloc((size_t)N_EDGES * 1024 * 2);
  unsigned short* h_b   = (unsigned short*)alloc((size_t)N_EDGES * 1024 * 2);
  unsigned short* nin   = (unsigned short*)alloc((size_t)N_EDGES * 544 * 2);
  float*          agg   = (float*)         alloc((size_t)N_NODES * 512 * 4);
  float*          cnt   = (float*)         alloc((size_t)N_NODES * 4);
  unsigned short* n2in  = (unsigned short*)alloc((size_t)N_NODES * 800 * 2);
  unsigned short* h3    = h_a;   // reuse (dead after node-MLP1 layer 0 consumed)

  // ---- prep: u reduction + weight transposes (bf16, K-padded) ----
  k_u_reduce<<<N_GRAPHS, U_RED, 0, stream>>>(u, w_dr, b_dr, ur);
  auto cvt = [&](const float* W, unsigned short* WT, int K, int N, int Kpad) {
    long long tot = (long long)N * Kpad;
    k_wT<<<(unsigned)((tot + 255) / 256), 256, 0, stream>>>(W, WT, K, N, Kpad);
  };
  cvt(e_w0, w0T,   275, 1024, 288);
  cvt(e_w1, w1T,  1024, 1024, 1024);
  cvt(e_w2, w2T,  1024, 1024, 1024);
  cvt(e_w3, w3T,  1024, 1024, 1024);
  cvt(e_wf, wfT,  1024,  512, 1024);
  cvt(n1w0, n1w0T, 521,  512, 544);
  cvt(n1w1, n1w1T, 512,  512, 512);
  cvt(n2w0, n2w0T, 777,  512, 800);

  // ---- gather/concat inputs ----
  k_build_e_in<<<N_EDGES, 288, 0, stream>>>(x, ea, ur, ei, bat, e_in);
  k_fill_n_in<<<N_EDGES, 32, 0, stream>>>(x, ei, nin);

  auto gemm = [&](const unsigned short* A, int lda, const unsigned short* BT, int ldb,
                  const float* bias, unsigned short* C, long long ldc, int cofs,
                  int M, int N, int K, int relu) {
    dim3 grid(N / 128, M / 128);
    k_gemm_bf16<<<grid, 256, 0, stream>>>(A, lda, BT, ldb, bias, C, ldc, cofs,
                                          M, N, K, relu);
  };

  // ---- edge MLP ----
  gemm(e_in, 288,  w0T, 288,  e_b0, h_a, 1024, 0, N_EDGES, 1024, 288,  1);
  gemm(h_a, 1024,  w1T, 1024, e_b1, h_b, 1024, 0, N_EDGES, 1024, 1024, 1);
  gemm(h_b, 1024,  w2T, 1024, e_b2, h_a, 1024, 0, N_EDGES, 1024, 1024, 1);
  gemm(h_a, 1024,  w3T, 1024, e_b3, h_b, 1024, 0, N_EDGES, 1024, 1024, 1);
  gemm(h_b, 1024,  wfT, 1024, e_bf, nin, 544,  9, N_EDGES, 512,  1024, 0); // e -> n_in[,9:521]

  // ---- node MLP 1 ----
  gemm(nin, 544, n1w0T, 544, n1b0, h_a, 512, 0, N_EDGES, 512, 544, 1);
  gemm(h_a, 512, n1w1T, 512, n1b1, h_b, 512, 0, N_EDGES, 512, 512, 1);   // h2 in h_b

  // ---- scatter-mean ----
  {
    long long na = (long long)N_NODES * 512;
    k_zero<<<(unsigned)((na + 255) / 256), 256, 0, stream>>>(agg, na);
    k_zero<<<(N_NODES + 255) / 256, 256, 0, stream>>>(cnt, N_NODES);
  }
  k_scatter<<<N_EDGES, 256, 0, stream>>>(h_b, ei, agg, cnt);
  k_build_n2_in<<<N_NODES, 256, 0, stream>>>(x, agg, cnt, ur, bat, n2in);

  // ---- node MLP 2 ----
  gemm(n2in, 800, n2w0T, 800, n2b0, h3, 512, 0, N_NODES, 512, 800, 1);
  k_final<<<N_NODES / 256, 256, 0, stream>>>(h3, n2w1, n2b1, out);
}
